// ColorDecoder_68238440399515
// MI455X (gfx1250) — matrix operand order repaired and checked by reading, not compiled, not run on hardware
//
#include <hip/hip_runtime.h>

typedef __attribute__((ext_vector_type(16))) _Float16 v16h;
typedef __attribute__((ext_vector_type(8)))  float    v8f;
typedef __attribute__((ext_vector_type(4)))  float    v4f;

#define BSZ   16
#define FDIM  256
#define HLO   64
#define PLO   4096        // 64*64
#define KCLS  21
#define HHI   512
#define WHI   512
#define PHI   (512*512)

#define BROW  272         // padded LDS row stride (halves): 544 B, 32B-aligned

// ---------------------------------------------------------------------------
// Kernel 1: logits[b][p][21] = fm[b,:,p] . conv_w[k,:] + bias  via WMMA f16
// One wave computes a 16(M=spatial) x 32(N=class, zero-padded) tile, K=256.
// conv_w is staged in LDS as f16 (rows 21..31 zeroed) so the inner loop has
// no predication: B fragments are contiguous aligned LDS reads.
// ---------------------------------------------------------------------------
__global__ __launch_bounds__(256) void k_logits_wmma(
    const float* __restrict__ fm, const float* __restrict__ cw,
    const float* __restrict__ cb, float* __restrict__ logits) {
  __shared__ _Float16 bsh[32 * BROW];           // ~17 KB

  int tid = threadIdx.x;                        // 256 threads == FDIM
#pragma unroll
  for (int n = 0; n < 32; ++n) {
    float v = (n < KCLS) ? cw[n * FDIM + tid] : 0.0f;
    bsh[n * BROW + tid] = (_Float16)v;
  }
  __syncthreads();

  int wave = (blockIdx.x * 256 + threadIdx.x) >> 5;   // 4096 waves total
  int lane = threadIdx.x & 31;
  int b    = wave >> 8;                               // 256 M-tiles per batch
  int p0   = (wave & 255) * 16;
  int M    = lane & 15;
  int half = lane >> 4;
  int khalf = half << 4;
  const float* fmb = fm + (size_t)b * FDIM * PLO + (p0 + M);

  v8f c0 = {}; v8f c1 = {};
#pragma unroll 2
  for (int kk = 0; kk < FDIM; kk += 32) {
    // A fragment: 16x32 f16, lane half h holds contiguous K = 16h..16h+15 (gather along F, stride PLO)
    v16h a;
#pragma unroll
    for (int j = 0; j < 16; ++j) {
      int K = khalf + j;
      a[j] = (_Float16)fmb[(size_t)(kk + K) * PLO];
    }
    // B fragments from LDS: lane's 16 K-values are contiguous & 32B-aligned
    v16h b0 = *(const v16h*)&bsh[M * BROW + kk + khalf];
    v16h b1 = *(const v16h*)&bsh[(16 + M) * BROW + kk + khalf];

    c0 = __builtin_amdgcn_wmma_f32_16x16x32_f16(false, a, false, b0,
                                                (short)0, c0, false, false);
    c1 = __builtin_amdgcn_wmma_f32_16x16x32_f16(false, a, false, b1,
                                                (short)0, c1, false, false);
  }
  // C layout: lanes 0-15 hold M=j, lanes 16-31 hold M=j+8; N = lane&15
  int N = lane & 15;
  float* lb = logits + (size_t)b * PLO * KCLS;
#pragma unroll
  for (int j = 0; j < 8; ++j) {
    int p = p0 + j + half * 8;
    lb[(size_t)p * KCLS + N] = c0[j] + cb[N];
    if (N < KCLS - 16) lb[(size_t)p * KCLS + 16 + N] = c1[j] + cb[16 + N];
  }
}

// ---------------------------------------------------------------------------
// Kernel 2: softmax over 21 classes; seg stored class-major [b][k][p]
// ---------------------------------------------------------------------------
__global__ __launch_bounds__(256) void k_softmax(
    const float* __restrict__ logits, float* __restrict__ seg) {
  int idx = blockIdx.x * 256 + threadIdx.x;           // b*4096 + p
  if (idx >= BSZ * PLO) return;
  int b = idx >> 12, p = idx & 4095;
  const float* l = logits + (size_t)idx * KCLS;
  float v[KCLS], m = -1e30f;
#pragma unroll
  for (int k = 0; k < KCLS; ++k) { v[k] = l[k]; m = fmaxf(m, v[k]); }
  float s = 0.0f;
#pragma unroll
  for (int k = 0; k < KCLS; ++k) { v[k] = __expf(v[k] - m); s += v[k]; }
  float inv = 1.0f / s;
  float* sb = seg + (size_t)b * KCLS * PLO + p;
#pragma unroll
  for (int k = 0; k < KCLS; ++k) sb[(size_t)k * PLO] = v[k] * inv;
}

// Adjoint bilinear (align_corners) weight: contribution of hi-res index i to
// lo-res index y.  scale = 63/511 exact rational.
__device__ __forceinline__ float adj_w(int i, int y) {
  int t  = i * 63;
  int y0 = t / 511;
  float f = (float)(t - y0 * 511) * (1.0f / 511.0f);
  return (y0 == y) ? (1.0f - f) : ((y0 == y - 1) ? f : 0.0f);
}

// ---------------------------------------------------------------------------
// Kernel 3a: rows adjoint:  tmp[b][c][y][j] = sum_i w(i,y) x[b,c,i,j]
// ---------------------------------------------------------------------------
__global__ __launch_bounds__(256) void k_down_rows(
    const float* __restrict__ x, float* __restrict__ tmp) {
  int idx = blockIdx.x * 256 + threadIdx.x;
  if (idx >= BSZ * 3 * HLO * WHI) return;
  int j  = idx % WHI;
  int y  = (idx / WHI) % HLO;
  int bc = idx / (WHI * HLO);
  const float* xr = x + (size_t)bc * (HHI * WHI);
  int ilo = (y <= 1) ? 0 : (511 * (y - 1) + 62) / 63;
  int ihi = (511 * (y + 1) + 62) / 63 - 1; if (ihi > HHI - 1) ihi = HHI - 1;
  float acc = 0.0f;
  for (int i = ilo; i <= ihi; ++i) acc += adj_w(i, y) * xr[(size_t)i * WHI + j];
  tmp[idx] = acc;
}

// ---------------------------------------------------------------------------
// Kernel 3b: cols adjoint:  xd[b][c][y][xq] = sum_j w(j,xq) tmp[b][c][y][j]
// ---------------------------------------------------------------------------
__global__ __launch_bounds__(256) void k_down_cols(
    const float* __restrict__ tmp, float* __restrict__ xd) {
  int idx = blockIdx.x * 256 + threadIdx.x;
  if (idx >= BSZ * 3 * PLO) return;
  int xq = idx % HLO;
  int y  = (idx / HLO) % HLO;
  int bc = idx / PLO;
  const float* tr = tmp + ((size_t)bc * HLO + y) * WHI;
  int jlo = (xq <= 1) ? 0 : (511 * (xq - 1) + 62) / 63;
  int jhi = (511 * (xq + 1) + 62) / 63 - 1; if (jhi > WHI - 1) jhi = WHI - 1;
  float acc = 0.0f;
  for (int j = jlo; j <= jhi; ++j) acc += adj_w(j, xq) * tr[j];
  xd[idx] = acc;
}

// ---------------------------------------------------------------------------
// Kernel 4: q[b][k][c] = (1/HW) * sum_p seg[b][k][p] * xd[b][c][p]
// ---------------------------------------------------------------------------
__global__ __launch_bounds__(128) void k_q(
    const float* __restrict__ seg, const float* __restrict__ xd,
    float* __restrict__ q) {
  int bk = blockIdx.x;                 // 336 blocks
  int b = bk / KCLS;
  const float* s  = seg + (size_t)bk * PLO;
  const float* xb = xd + (size_t)b * 3 * PLO;
  float a0 = 0, a1 = 0, a2 = 0;
  for (int p = threadIdx.x; p < PLO; p += 128) {
    float sv = s[p];
    a0 += sv * xb[p];
    a1 += sv * xb[PLO + p];
    a2 += sv * xb[2 * PLO + p];
  }
  __shared__ float red[3][128];
  int t = threadIdx.x;
  red[0][t] = a0; red[1][t] = a1; red[2][t] = a2;
  __syncthreads();
  for (int off = 64; off > 0; off >>= 1) {
    if (t < off) {
      red[0][t] += red[0][t + off];
      red[1][t] += red[1][t + off];
      red[2][t] += red[2][t + off];
    }
    __syncthreads();
  }
  if (t == 0) {
    const float inv = 1.0f / (float)PHI;
    q[bk * 3 + 0] = red[0][0] * inv;
    q[bk * 3 + 1] = red[1][0] * inv;
    q[bk * 3 + 2] = red[2][0] * inv;
  }
}

// ---------------------------------------------------------------------------
// Kernel 5 (bandwidth-bound): attn[b][k][p] = sum_c x[b][c][p] * q[b][k][c]
// b128 loads of x, 21 non-temporal b128 stores per thread (352 MB streamed).
// ---------------------------------------------------------------------------
__global__ __launch_bounds__(256) void k_attn(
    const float* __restrict__ x, const float* __restrict__ q,
    float* __restrict__ out) {
  int b  = blockIdx.x >> 8;                  // 256 chunks of 1024 px per batch
  int p4 = (blockIdx.x & 255) * 1024 + threadIdx.x * 4;
  const float* xb = x + (size_t)b * 3 * PHI;
  v4f x0 = *(const v4f*)(xb + p4);
  v4f x1 = *(const v4f*)(xb + PHI + p4);
  v4f x2 = *(const v4f*)(xb + 2 * PHI + p4);

  __shared__ float qs[KCLS * 3];
  if (threadIdx.x < KCLS * 3) qs[threadIdx.x] = q[b * KCLS * 3 + threadIdx.x];
  __syncthreads();

  float* ob = out + (size_t)b * KCLS * PHI + p4;
#pragma unroll
  for (int k = 0; k < KCLS; ++k) {
    float q0 = qs[k * 3], q1 = qs[k * 3 + 1], q2 = qs[k * 3 + 2];
    v4f r = x0 * q0 + x1 * q1 + x2 * q2;
    __builtin_nontemporal_store(r, (v4f*)(ob + (size_t)k * PHI));
  }
}

// ---------------------------------------------------------------------------
extern "C" void kernel_launch(void* const* d_in, const int* in_sizes, int n_in,
                              void* d_out, int out_size, void* d_ws,
                              size_t ws_size, hipStream_t stream) {
  const float* fm = (const float*)d_in[0];   // [16,256,64,64]
  const float* x  = (const float*)d_in[1];   // [16,3,512,512]
  const float* cw = (const float*)d_in[2];   // [21,256]
  const float* cb = (const float*)d_in[3];   // [21]
  float* out = (float*)d_out;                // [16,21,512,512]

  // Workspace layout (floats): ~18.1 MB total
  float* wsf    = (float*)d_ws;
  size_t nLog   = (size_t)BSZ * PLO * KCLS;        // 1,376,256
  size_t nTmp   = (size_t)BSZ * 3 * HLO * WHI;     // 1,572,864
  size_t nXd    = (size_t)BSZ * 3 * PLO;           //   196,608
  float* logits = wsf;
  float* seg    = logits + nLog;
  float* tmp    = seg + nLog;
  float* xd     = tmp + nTmp;
  float* q      = xd + nXd;

  // 1) logits GEMM via WMMA: 4096 waves = 512 blocks x 256 thr
  k_logits_wmma<<<512, 256, 0, stream>>>(fm, cw, cb, logits);
  // 2) softmax 64x64x21
  k_softmax<<<(BSZ * PLO + 255) / 256, 256, 0, stream>>>(logits, seg);
  // 3) adjoint bilinear downsample of x (separable)
  k_down_rows<<<(BSZ * 3 * HLO * WHI + 255) / 256, 256, 0, stream>>>(x, tmp);
  k_down_cols<<<(BSZ * 3 * PLO + 255) / 256, 256, 0, stream>>>(tmp, xd);
  // 4) q[b,k,c]
  k_q<<<BSZ * KCLS, 128, 0, stream>>>(seg, xd, q);
  // 5) streamed output
  k_attn<<<BSZ * 256, 256, 0, stream>>>(x, q, out);
}
